// PureSwinSegmentation_7799660609926
// MI455X (gfx1250) — matrix operand order repaired and checked by reading, not compiled, not run on hardware
//
#include <hip/hip_runtime.h>

typedef __bf16 bf16;
typedef __attribute__((ext_vector_type(16))) __bf16 v16bf;
typedef __attribute__((ext_vector_type(8)))  float  v8f;

#define TOK 200704L

// ---------------------------------------------------------------------------
// CDNA5 async global -> LDS copy (16B per lane), tracked by ASYNCcnt
// ---------------------------------------------------------------------------
__device__ __forceinline__ void async_copy16(uint32_t lds_off, const void* gptr) {
    asm volatile("global_load_async_to_lds_b128 %0, %1, off"
                 :: "v"(lds_off), "v"(gptr) : "memory");
}
__device__ __forceinline__ void wait_async0() {
    asm volatile("s_wait_asynccnt 0x0" ::: "memory");
}
__device__ __forceinline__ void wait_async4() {
    asm volatile("s_wait_asynccnt 0x4" ::: "memory");
}
__device__ __forceinline__ uint32_t lds_off32(const void* p) {
    return (uint32_t)(uintptr_t)p;  // flat->LDS: addr[31:0] is the LDS offset
}

// ---------------------------------------------------------------------------
// fp32 -> bf16 weight convert
// ---------------------------------------------------------------------------
__global__ void cvt_kernel(const float* __restrict__ s, bf16* __restrict__ d, int n) {
    int i = blockIdx.x * 256 + threadIdx.x;
    if (i < n) d[i] = (bf16)s[i];
}

// ---------------------------------------------------------------------------
// LayerNorm over C=256, one wave (32 lanes) per token, 8 tokens per block
// ---------------------------------------------------------------------------
__global__ __launch_bounds__(256) void ln_kernel(const float* __restrict__ in,
                                                 const float* __restrict__ g,
                                                 const float* __restrict__ b,
                                                 bf16* __restrict__ out) {
    const int lane = threadIdx.x & 31;
    const long t = (long)blockIdx.x * 8 + (threadIdx.x >> 5);
    const float* row = in + t * 256;
    float v[8], s1 = 0.f, s2 = 0.f;
    for (int i = 0; i < 8; ++i) {
        v[i] = row[i * 32 + lane];
        s1 += v[i];
        s2 += v[i] * v[i];
    }
    for (int m = 16; m >= 1; m >>= 1) {
        s1 += __shfl_xor(s1, m, 32);
        s2 += __shfl_xor(s2, m, 32);
    }
    const float mu = s1 * (1.0f / 256.0f);
    const float var = s2 * (1.0f / 256.0f) - mu * mu;
    const float rs = rsqrtf(var + 1e-5f);
    for (int i = 0; i < 8; ++i) {
        const int c = i * 32 + lane;
        out[t * 256 + c] = (bf16)((v[i] - mu) * rs * g[c] + b[c]);
    }
}

// ---------------------------------------------------------------------------
// Generic bf16 WMMA GEMM: D[M,N] = A[M,K] @ Bw[N,K]^T + bias, with epilogues:
// EPI 0: store bf16          EPI 1: GELU(exact) -> bf16
// EPI 2: +X residual -> f32  EPI 3: accumulate into f32 Df
// Block = 64x64 tile, 128 threads = 4 waves, each wave 32x32 (2x2 WMMA frags)
// Double-buffered async global->LDS staging (ASYNCcnt pipelined).
// ---------------------------------------------------------------------------
template <int EPI>
__global__ __launch_bounds__(128) void gemm_kernel(const bf16* __restrict__ A,
                                                   const bf16* __restrict__ Bw,
                                                   const float* __restrict__ bias,
                                                   bf16* __restrict__ Dbf,
                                                   float* __restrict__ Df,
                                                   const float* __restrict__ X,
                                                   int M, int N, int K) {
    __shared__ __align__(16) bf16 As[2][64][32];
    __shared__ __align__(16) bf16 Bs[2][64][32];

    const int tid = threadIdx.x;
    const int lane = tid & 31, wave = tid >> 5;
    const int wm = wave >> 1, wn = wave & 1;
    const long m0 = (long)blockIdx.y * 64;
    const long n0 = (long)blockIdx.x * 64;

    const int la = (lane < 16) ? 0 : 8;    // A-frag K base (elems 0..7)
    const int lb = (lane < 16) ? 0 : 8;    // B-frag K base (elems 0..7)
    const int mloc = lane & 15;

    // this thread's two staging chunks (16B each per matrix)
    const int row0 = tid >> 2,          off0 = (tid & 3) * 8;
    const int row1 = (tid + 128) >> 2,  off1 = ((tid + 128) & 3) * 8;

    v8f acc[2][2] = {};

    auto stage = [&](int s, int k1) {
        async_copy16(lds_off32(&As[s][row0][off0]), &A[(m0 + row0) * K + k1 + off0]);
        async_copy16(lds_off32(&Bs[s][row0][off0]), &Bw[(n0 + row0) * K + k1 + off0]);
        async_copy16(lds_off32(&As[s][row1][off1]), &A[(m0 + row1) * K + k1 + off1]);
        async_copy16(lds_off32(&Bs[s][row1][off1]), &Bw[(n0 + row1) * K + k1 + off1]);
    };

    const int nk = K >> 5;
    stage(0, 0);
    for (int kb = 0; kb < nk; ++kb) {
        const int s = kb & 1;
        if (kb + 1 < nk) {
            stage(s ^ 1, (kb + 1) * 32);  // prefetch next tile into other buffer
            wait_async4();                // first 4 (this tile) complete in-order
        } else {
            wait_async0();
        }
        __syncthreads();

        v16bf af[2], bfm[2];
        for (int i = 0; i < 2; ++i) {
            const int m = wm * 32 + i * 16 + mloc;
            union { v16bf v; float4 f[2]; } u;
            u.f[0] = *(const float4*)&As[s][m][la];
            u.f[1] = *(const float4*)&As[s][m][la + 16];
            af[i] = u.v;
            const int n = wn * 32 + i * 16 + mloc;
            union { v16bf v; float4 f[2]; } w;
            w.f[0] = *(const float4*)&Bs[s][n][lb];
            w.f[1] = *(const float4*)&Bs[s][n][lb + 16];
            bfm[i] = w.v;
        }
        for (int i = 0; i < 2; ++i)
            for (int j = 0; j < 2; ++j)
                acc[i][j] = __builtin_amdgcn_wmma_f32_16x16x32_bf16(
                    false, af[i], false, bfm[j], (short)0, acc[i][j], false, false);
        __syncthreads();
    }

    const int rbase = (lane < 16) ? 0 : 8;
    for (int i = 0; i < 2; ++i)
        for (int j = 0; j < 2; ++j) {
            const long col = n0 + wn * 32 + j * 16 + mloc;
            const float bc = bias[col];
            for (int r = 0; r < 8; ++r) {
                const long row = m0 + wm * 32 + i * 16 + rbase + r;
                const long idx = row * N + col;
                float v = acc[i][j][r] + bc;
                if (EPI == 0) {
                    Dbf[idx] = (bf16)v;
                } else if (EPI == 1) {
                    const float gch = 0.5f * v * (1.0f + erff(v * 0.70710678118654752f));
                    Dbf[idx] = (bf16)gch;
                } else if (EPI == 2) {
                    Df[idx] = v + X[idx];
                } else {
                    Df[idx] += v;
                }
            }
        }
}

// ---------------------------------------------------------------------------
// Windowed attention: one block per (window, head); 128 threads = 4 waves.
// S = Q K^T * scale + bias, softmax over 49 keys, O = P V.  49 padded -> 64.
// Q/K tiles staged via async global->LDS.
// ---------------------------------------------------------------------------
__global__ __launch_bounds__(128) void attn_kernel(const bf16* __restrict__ qkv,
                                                   const float* __restrict__ bias_table,
                                                   const int* __restrict__ rel_index,
                                                   bf16* __restrict__ obuf) {
    __shared__ __align__(16) bf16 qs[64][32];
    __shared__ __align__(16) bf16 ks[64][32];
    __shared__ __align__(16) bf16 vst[32][64];  // vst[dim][key]
    __shared__ __align__(16) float Ssm[64][65];
    __shared__ __align__(16) bf16 Ps[64][64];
    __shared__ float biasL[49 * 49];

    const int tid = threadIdx.x;
    const int head = blockIdx.x & 7;
    const int win = blockIdx.x >> 3;
    const int bimg = win >> 10;
    const int wy = (win >> 5) & 31;
    const int wx = win & 31;
    const int y0 = wy * 7, x0 = wx * 7;
    const long hb = (long)head * 32;

    auto tok = [&](int r) -> long {
        return ((long)(bimg * 224 + y0 + r / 7)) * 224 + (x0 + r % 7);
    };

    // async-stage valid Q/K rows: 2 mats * 49 rows * 4 chunks = 392 chunks
    for (int idx = tid; idx < 392; idx += 128) {
        const int mat = idx / 196;
        const int rem = idx - mat * 196;
        const int r = rem >> 2, ck = (rem & 3) * 8;
        bf16(*dst)[32] = mat ? ks : qs;
        async_copy16(lds_off32(&dst[r][ck]), qkv + tok(r) * 768 + hb + mat * 256 + ck);
    }
    // zero-pad rows 49..63 of qs/ks (plain DS stores)
    if (tid < 30) {
        const int mat = tid / 15;
        const int r = 49 + tid - mat * 15;
        bf16(*dst)[32] = mat ? ks : qs;
        const float4 z = {};
        float4* d = (float4*)&dst[r][0];
        d[0] = z; d[1] = z; d[2] = z; d[3] = z;
    }
    // stage relative-position bias for this head
    for (int i = tid; i < 49 * 49; i += 128)
        biasL[i] = bias_table[rel_index[i] * 8 + head];
    // stage V transposed
    for (int idx = tid; idx < 64 * 32; idx += 128) {
        const int r = idx >> 5, c = idx & 31;
        vst[c][r] = (r < 49) ? qkv[tok(r) * 768 + hb + 512 + c] : (bf16)0.0f;
    }
    wait_async0();
    __syncthreads();

    const int lane = tid & 31, wave = tid >> 5;
    const int la = (lane < 16) ? 0 : 8;
    const int lb = (lane < 16) ? 0 : 8;
    const int nloc = lane & 15;
    const int rb = wave * 16 + ((lane < 16) ? 0 : 8);

    // S = Q K^T : wave handles M-tile rows [16*wave, 16*wave+16)
    {
        union { v16bf v; float4 f[2]; } u;
        const int m = wave * 16 + nloc;
        u.f[0] = *(const float4*)&qs[m][la];
        u.f[1] = *(const float4*)&qs[m][la + 16];
        for (int nt = 0; nt < 4; ++nt) {
            union { v16bf v; float4 f[2]; } w;
            const int n = nt * 16 + nloc;
            w.f[0] = *(const float4*)&ks[n][lb];
            w.f[1] = *(const float4*)&ks[n][lb + 16];
            v8f sacc = {};
            sacc = __builtin_amdgcn_wmma_f32_16x16x32_bf16(
                false, u.v, false, w.v, (short)0, sacc, false, false);
            for (int r = 0; r < 8; ++r) Ssm[rb + r][nt * 16 + nloc] = sacc[r];
        }
    }
    __syncthreads();

    // softmax over 49 valid keys (rows handled by threads 0..63)
    const float scale = 0.17677669529663687f;  // 32^-0.5
    if (tid < 64) {
        const int r = tid;
        if (r < 49) {
            float mx = -1e30f;
            for (int c = 0; c < 49; ++c) {
                const float s = Ssm[r][c] * scale + biasL[r * 49 + c];
                Ssm[r][c] = s;
                mx = fmaxf(mx, s);
            }
            float sum = 0.f;
            for (int c = 0; c < 49; ++c) {
                const float e = expf(Ssm[r][c] - mx);
                Ssm[r][c] = e;
                sum += e;
            }
            const float inv = 1.0f / sum;
            for (int c = 0; c < 49; ++c) Ps[r][c] = (bf16)(Ssm[r][c] * inv);
            for (int c = 49; c < 64; ++c) Ps[r][c] = (bf16)0.0f;
        } else {
            for (int c = 0; c < 64; ++c) Ps[r][c] = (bf16)0.0f;
        }
    }
    __syncthreads();

    // O = P V : 2 K-steps of 32, 2 N-tiles of 16 (d = 32)
    v8f oacc[2] = {};
    for (int kk = 0; kk < 2; ++kk) {
        union { v16bf v; float4 f[2]; } u;
        const int m = wave * 16 + nloc;
        u.f[0] = *(const float4*)&Ps[m][kk * 32 + la];
        u.f[1] = *(const float4*)&Ps[m][kk * 32 + la + 16];
        for (int nt = 0; nt < 2; ++nt) {
            union { v16bf v; float4 f[2]; } w;
            const int n = nt * 16 + nloc;
            w.f[0] = *(const float4*)&vst[n][kk * 32 + lb];
            w.f[1] = *(const float4*)&vst[n][kk * 32 + lb + 16];
            oacc[nt] = __builtin_amdgcn_wmma_f32_16x16x32_bf16(
                false, u.v, false, w.v, (short)0, oacc[nt], false, false);
        }
    }
    // write O token-major (fuses window_reverse)
    for (int nt = 0; nt < 2; ++nt) {
        for (int r = 0; r < 8; ++r) {
            const int row = rb + r;
            if (row < 49) {
                const long t = tok(row);
                obuf[t * 256 + hb + nt * 16 + nloc] = (bf16)oacc[nt][r];
            }
        }
    }
}

// ---------------------------------------------------------------------------
// launch
// ---------------------------------------------------------------------------
extern "C" void kernel_launch(void* const* d_in, const int* in_sizes, int n_in,
                              void* d_out, int out_size, void* d_ws, size_t ws_size,
                              hipStream_t stream) {
    (void)in_sizes; (void)n_in; (void)out_size; (void)ws_size;
    const float* x          = (const float*)d_in[0];
    const float* n1g        = (const float*)d_in[1];
    const float* n1b        = (const float*)d_in[2];
    const float* qkv_w      = (const float*)d_in[3];
    const float* qkv_b      = (const float*)d_in[4];
    const float* bias_table = (const float*)d_in[5];
    const float* proj_w     = (const float*)d_in[6];
    const float* proj_b     = (const float*)d_in[7];
    const float* n2g        = (const float*)d_in[8];
    const float* n2b        = (const float*)d_in[9];
    const float* fc1_w      = (const float*)d_in[10];
    const float* fc1_b      = (const float*)d_in[11];
    const float* fc2_w      = (const float*)d_in[12];
    const float* fc2_b      = (const float*)d_in[13];
    const int*   rel_index  = (const int*)d_in[14];
    float* out = (float*)d_out;

    char* ws = (char*)d_ws;
    bf16* qkvbuf = (bf16*)ws;                              // TOK x 768 bf16
    bf16* obuf   = (bf16*)(ws + TOK * 768 * 2);            // TOK x 256 bf16
    bf16* xn     = (bf16*)(ws + TOK * 1024 * 2);           // TOK x 256 bf16 (xn, then h2)
    bf16* mlp    = (bf16*)ws;                              // TOK x 1024 bf16 (reuses qkv+o)
    bf16* wq = (bf16*)(ws + TOK * 1024 * 2 + TOK * 256 * 2);
    bf16* wp = wq + 768 * 256;
    bf16* w1 = wp + 256 * 256;
    bf16* w2 = w1 + 1024 * 256;

    cvt_kernel<<<(768 * 256 + 255) / 256, 256, 0, stream>>>(qkv_w, wq, 768 * 256);
    cvt_kernel<<<(256 * 256 + 255) / 256, 256, 0, stream>>>(proj_w, wp, 256 * 256);
    cvt_kernel<<<(1024 * 256 + 255) / 256, 256, 0, stream>>>(fc1_w, w1, 1024 * 256);
    cvt_kernel<<<(1024 * 256 + 255) / 256, 256, 0, stream>>>(fc2_w, w2, 256 * 1024);

    // LN1 -> xn (bf16)
    ln_kernel<<<TOK / 8, 256, 0, stream>>>(x, n1g, n1b, xn);
    // QKV: xn @ qkv_w^T + b -> qkvbuf (bf16)
    gemm_kernel<0><<<dim3(12, TOK / 64), 128, 0, stream>>>(xn, wq, qkv_b, qkvbuf,
                                                           nullptr, nullptr, (int)TOK, 768, 256);
    // windowed attention -> obuf (token-major, bf16)
    attn_kernel<<<4096 * 8, 128, 0, stream>>>(qkvbuf, bias_table, rel_index, obuf);
    // proj + residual: out = x + obuf @ proj_w^T + b   (f32)
    gemm_kernel<2><<<dim3(4, TOK / 64), 128, 0, stream>>>(obuf, wp, proj_b, nullptr,
                                                          out, x, (int)TOK, 256, 256);
    // LN2 -> h2 (bf16, reuses xn region)
    ln_kernel<<<TOK / 8, 256, 0, stream>>>(out, n2g, n2b, xn);
    // fc1 + GELU -> mlp (bf16, reuses qkv+o regions)
    gemm_kernel<1><<<dim3(16, TOK / 64), 128, 0, stream>>>(xn, w1, fc1_b, mlp,
                                                           nullptr, nullptr, (int)TOK, 1024, 256);
    // fc2 accumulate into out
    gemm_kernel<3><<<dim3(4, TOK / 64), 128, 0, stream>>>(mlp, w2, fc2_b, nullptr,
                                                          out, nullptr, (int)TOK, 256, 1024);
}
